// DilatedRNN_34668976013370
// MI455X (gfx1250) — matrix operand order repaired and checked by reading, not compiled, not run on hardware
//
#include <hip/hip_runtime.h>
#include <math.h>

#define B_     32
#define T_     2048
#define H_     256
#define DEPTH_ 4
#define LDSPAD 264   // 256 + 8 bf16 pad to break LDS bank conflicts

typedef __bf16 bf16_t;
typedef __attribute__((ext_vector_type(16))) __bf16 v16bf;
typedef __attribute__((ext_vector_type(8)))  __bf16 v8bf;
typedef __attribute__((ext_vector_type(8)))  float  v8f;

// ---------------------------------------------------------------------------
// WMMA fragment loaders (wave32, 16x16x32 bf16).
// A (16x32, MxK), row-major source with leading dim ld:
//   lane<16 : m=lane,   elems 0..7 -> K k0+0..7,   elems 8..15 -> K k0+16..23
//   lane>=16: m=lane-16,elems 0..7 -> K k0+8..15,  elems 8..15 -> K k0+24..31
// ---------------------------------------------------------------------------
__device__ __forceinline__ v16bf load_a_frag(const bf16_t* __restrict__ src,
                                             int ld, int k0) {
  const int lane = threadIdx.x & 31;
  const int m    = lane & 15;
  const int hi   = lane >> 4;           // 0 or 1
  const bf16_t* p = src + (long)m * ld + k0 + 8 * hi;
  v8bf lo = *(const v8bf*)(p);
  v8bf up = *(const v8bf*)(p + 16);
  v16bf a;
#pragma unroll
  for (int e = 0; e < 8; ++e) { a[e] = lo[e]; a[8 + e] = up[e]; }
  return a;
}

// B (32x16, KxN) from a pre-transposed weight Wt[n][k] (contiguous in k):
//   lane<16 : n=lane,    elems 0..7 -> K k0+0..7,   elems 8..15 -> K k0+16..23
//   lane>=16: n=lane-16, elems 0..7 -> K k0+8..15,  elems 8..15 -> K k0+24..31
__device__ __forceinline__ v16bf load_b_frag_t(const bf16_t* __restrict__ wt,
                                               int ldk, int n0, int k0) {
  const int lane = threadIdx.x & 31;
  const int n    = lane & 15;
  const int hi   = lane >> 4;
  const bf16_t* p = wt + (long)(n0 + n) * ldk + k0 + 8 * hi;
  v8bf b0 = *(const v8bf*)(p);
  v8bf b1 = *(const v8bf*)(p + 16);
  v16bf b;
#pragma unroll
  for (int e = 0; e < 8; ++e) { b[e] = b0[e]; b[8 + e] = b1[e]; }
  return b;
}

__device__ __forceinline__ v8f wmma_bf16(v16bf a, v16bf b, v8f c) {
  return __builtin_amdgcn_wmma_f32_16x16x32_bf16(
      false, a, false, b, (short)0, c, false, false);
}

// ---------------------------------------------------------------------------
// fp32 -> bf16 elementwise convert
// ---------------------------------------------------------------------------
__global__ void cvt_f32_bf16(const float* __restrict__ in,
                             bf16_t* __restrict__ out, int n) {
  int i = blockIdx.x * blockDim.x + threadIdx.x;
  if (i < n) out[i] = (bf16_t)in[i];
}

// Convert fp32 weights [depth][k][n] into transposed bf16 [depth][n][k]
__global__ void cvt_transpose_w(const float* __restrict__ W,
                                bf16_t* __restrict__ Wt, int total) {
  int i = blockIdx.x * blockDim.x + threadIdx.x;
  if (i >= total) return;
  int l = i / (H_ * H_);
  int r = i % (H_ * H_);
  int k = r / H_;
  int n = r % H_;
  Wt[(long)l * H_ * H_ + (long)n * H_ + k] = (bf16_t)W[i];
}

// ---------------------------------------------------------------------------
// Z = Xbf @ Wx + b   (M = B*T rows, K = N = 256)
// grid.x = M/16 ; block = 256 threads = 8 waves, wave w owns cols [32w,32w+32)
// ---------------------------------------------------------------------------
__global__ __launch_bounds__(256)
void gemm_z(const bf16_t* __restrict__ X, const bf16_t* __restrict__ WxT,
            const float* __restrict__ bias, float* __restrict__ Z) {
  const int m0   = blockIdx.x * 16;
  const int wave = threadIdx.x >> 5;
  const int lane = threadIdx.x & 31;
  const int n0   = wave * 32;

  v8f acc0; v8f acc1;
#pragma unroll
  for (int i = 0; i < 8; ++i) { acc0[i] = 0.f; acc1[i] = 0.f; }

  const bf16_t* Xrow = X + (long)m0 * H_;
#pragma unroll
  for (int kc = 0; kc < 8; ++kc) {
    const int k0 = kc * 32;
    v16bf a  = load_a_frag(Xrow, H_, k0);
    v16bf b0 = load_b_frag_t(WxT, H_, n0,      k0);
    v16bf b1 = load_b_frag_t(WxT, H_, n0 + 16, k0);
    acc0 = wmma_bf16(a, b0, acc0);
    acc1 = wmma_bf16(a, b1, acc1);
  }

  const int hi = lane >> 4, nl = lane & 15;
  const int c0 = n0 + nl, c1 = n0 + 16 + nl;
  const float bb0 = bias[c0], bb1 = bias[c1];
#pragma unroll
  for (int i = 0; i < 8; ++i) {
    int m = i + 8 * hi;
    long row = (long)(m0 + m) * H_;
    Z[row + c0] = acc0[i] + bb0;
    Z[row + c1] = acc1[i] + bb1;
  }
}

// ---------------------------------------------------------------------------
// Dilated scan. One block per group of 16 independent streams.
// h_t = tanh(Z_t + h_{t-1} @ Wh), h in LDS (bf16), Wh B-fragments in VGPRs.
// Writes masked fp32 to out, unmasked bf16 to Xbf (next layer's input).
// ---------------------------------------------------------------------------
__global__ __launch_bounds__(256, 1)
void scan_layer(const float* __restrict__ Z, const bf16_t* __restrict__ WhT,
                const int* __restrict__ seq_lens, float* __restrict__ out,
                bf16_t* __restrict__ Xbf, int layer, int dil) {
  __shared__ bf16_t lds_h[16][LDSPAD];

  const int wave = threadIdx.x >> 5;
  const int lane = threadIdx.x & 31;
  const int n0   = wave * 32;
  const int hi   = lane >> 4, nl = lane & 15;
  const int c0   = n0 + nl, c1 = n0 + 16 + nl;

  // Wh fragments for this wave's 32 columns: 16 x v16bf, register resident.
  v16bf bw0[8], bw1[8];
#pragma unroll
  for (int kc = 0; kc < 8; ++kc) {
    bw0[kc] = load_b_frag_t(WhT, H_, n0,      kc * 32);
    bw1[kc] = load_b_frag_t(WhT, H_, n0 + 16, kc * 32);
  }

  // zero initial hidden state
  for (int i = threadIdx.x; i < 16 * LDSPAD; i += 256)
    (&lds_h[0][0])[i] = (bf16_t)0.0f;
  __syncthreads();

  // per-lane stream metadata for the 8 accumulator rows (m = i + 8*hi)
  int zbase[8], bidx_[8], r_[8], slen_[8];
#pragma unroll
  for (int i = 0; i < 8; ++i) {
    int s = i + 8 * hi;
    int g = blockIdx.x * 16 + s;       // global stream id
    int bb = g / dil, rr = g % dil;
    bidx_[i] = bb; r_[i] = rr;
    zbase[i] = bb * T_ + rr;           // Z row index at t=0
    slen_[i] = seq_lens[bb];
  }

  const int steps = T_ / dil;          // T divisible by 1,2,4,8

  for (int t = 0; t < steps; ++t) {
    v8f acc0, acc1;
#pragma unroll
    for (int i = 0; i < 8; ++i) {
      long row = (long)(zbase[i] + t * dil) * H_;
      acc0[i] = Z[row + c0];
      acc1[i] = Z[row + c1];
      if (t + 1 < steps) {             // prefetch next step's Z rows
        long nrow = (long)(zbase[i] + (t + 1) * dil) * H_;
        __builtin_prefetch(&Z[nrow + c0], 0, 0);
      }
    }

#pragma unroll
    for (int kc = 0; kc < 8; ++kc) {
      v16bf a = load_a_frag(&lds_h[0][0], LDSPAD, kc * 32);
      acc0 = wmma_bf16(a, bw0[kc], acc0);
      acc1 = wmma_bf16(a, bw1[kc], acc1);
    }
    __syncthreads();                   // all reads of h_{t-1} complete

#pragma unroll
    for (int i = 0; i < 8; ++i) {
      int   m    = i + 8 * hi;
      float v0   = tanhf(acc0[i]);
      float v1   = tanhf(acc1[i]);
      int   tpos = t * dil + r_[i];
      bool  ok   = tpos < slen_[i];
      long  ob   = ((long)(bidx_[i] * DEPTH_ + layer) * T_ + tpos) * H_;
      out[ob + c0] = ok ? v0 : 0.0f;
      out[ob + c1] = ok ? v1 : 0.0f;
      long  xr   = (long)(bidx_[i] * T_ + tpos) * H_;
      Xbf[xr + c0] = (bf16_t)v0;       // unmasked, feeds next layer
      Xbf[xr + c1] = (bf16_t)v1;
      lds_h[m][c0] = (bf16_t)v0;
      lds_h[m][c1] = (bf16_t)v1;
    }
    __syncthreads();                   // h_t visible before next step
  }
}

// ---------------------------------------------------------------------------
extern "C" void kernel_launch(void* const* d_in, const int* in_sizes, int n_in,
                              void* d_out, int out_size, void* d_ws, size_t ws_size,
                              hipStream_t stream) {
  const float* x   = (const float*)d_in[0];   // [B,T,H]
  const float* Wx  = (const float*)d_in[1];   // [DEPTH,H,H]
  const float* Wh  = (const float*)d_in[2];   // [DEPTH,H,H]
  const float* b   = (const float*)d_in[3];   // [DEPTH,H]
  const int*   sl  = (const int*)d_in[4];     // [B]
  float*       out = (float*)d_out;           // [B,DEPTH,T,H]

  const long NROW = (long)B_ * T_;            // 65536
  const long NXH  = NROW * H_;                // 16,777,216 elements

  char* ws = (char*)d_ws;
  bf16_t* Xbf = (bf16_t*)ws;                       ws += NXH * sizeof(bf16_t);   // 32 MB
  float*  Z   = (float*)ws;                        ws += NXH * sizeof(float);    // 64 MB
  bf16_t* WxT = (bf16_t*)ws;                       ws += (long)DEPTH_ * H_ * H_ * sizeof(bf16_t);
  bf16_t* WhT = (bf16_t*)ws;

  // 1. convert x -> bf16
  {
    int n = (int)NXH;
    cvt_f32_bf16<<<(n + 255) / 256, 256, 0, stream>>>(x, Xbf, n);
  }
  // 2. convert + transpose weights -> bf16
  {
    int tw = DEPTH_ * H_ * H_;
    cvt_transpose_w<<<(tw + 255) / 256, 256, 0, stream>>>(Wx, WxT, tw);
    cvt_transpose_w<<<(tw + 255) / 256, 256, 0, stream>>>(Wh, WhT, tw);
  }

  // 3. per layer: dense GEMM for Z, then dilated scan
  for (int l = 0; l < DEPTH_; ++l) {
    const int dil = 1 << l;
    gemm_z<<<(int)(NROW / 16), 256, 0, stream>>>(
        Xbf, WxT + (long)l * H_ * H_, b + (long)l * H_, Z);
    const int groups = (B_ * dil) / 16;       // 2,4,8,16 blocks
    scan_layer<<<groups, 256, 0, stream>>>(
        Z, WhT + (long)l * H_ * H_, sl, out, Xbf, l, dil);
  }
}
